// PointNeXtSetAbstraction_38886633898302
// MI455X (gfx1250) — matrix operand order repaired and checked by reading, not compiled, not run on hardware
//
#include <hip/hip_runtime.h>

// Problem constants (from reference setup_inputs)
#define NB   4          // batches
#define NP   8192       // points
#define NS   2048       // npoint (fps samples)
#define NK   32         // nsample (ball query)
#define CINP 32         // 3 rel-xyz + 29 feature channels
#define CMID 64         // COUT
#define NDF  29         // raw point feature channels
#define NCOL (NB*NK*NS) // 262144 GEMM columns
#define R2   0.01f      // radius^2
#define BNEPS 1e-5f

typedef _Float16 v16h __attribute__((ext_vector_type(16)));
typedef _Float16 v8h  __attribute__((ext_vector_type(8)));
typedef float    v8f  __attribute__((ext_vector_type(8)));

// ---------------------------------------------------------------------------
// Farthest point sampling: 1 block per batch, 1024 threads, 8 points/thread.
// Sequential over 2048 iterations; argmax with lowest-index tie-break to
// match jnp.argmax. Matches lax.scan semantics: fidx[j] = far BEFORE update.
// ---------------------------------------------------------------------------
__global__ __launch_bounds__(1024) void fps_kernel(const float* __restrict__ xyz,
                                                   int* __restrict__ fidx) {
  const int b = blockIdx.x, tid = threadIdx.x;
  const float* xb = xyz + b * 3 * NP;
  float px[8], py[8], pz[8], dist[8];
#pragma unroll
  for (int i = 0; i < 8; i++) {
    int p = tid + i * 1024;
    px[i] = xb[p]; py[i] = xb[NP + p]; pz[i] = xb[2 * NP + p];
    dist[i] = 1e10f;
  }
  __shared__ float s_val[32];
  __shared__ int   s_idx[32];
  __shared__ float s_c[3];
  __shared__ int   s_far;
  int far = 0;
  for (int j = 0; j < NS; j++) {
    if (tid == 0) {
      fidx[b * NS + j] = far;
      s_c[0] = xb[far]; s_c[1] = xb[NP + far]; s_c[2] = xb[2 * NP + far];
    }
    __syncthreads();
    float cx = s_c[0], cy = s_c[1], cz = s_c[2];
    float best = -1.0f; int bi = 0;
#pragma unroll
    for (int i = 0; i < 8; i++) {
      float dx = px[i] - cx, dy = py[i] - cy, dz = pz[i] - cz;
      float d = dx * dx + dy * dy + dz * dz;
      d = fminf(dist[i], d);
      dist[i] = d;
      if (d > best) { best = d; bi = tid + i * 1024; }
    }
    // wave32 reduce
#pragma unroll
    for (int off = 16; off > 0; off >>= 1) {
      float ov = __shfl_down(best, off, 32);
      int   oi = __shfl_down(bi,   off, 32);
      if (ov > best || (ov == best && oi < bi)) { best = ov; bi = oi; }
    }
    if ((tid & 31) == 0) { s_val[tid >> 5] = best; s_idx[tid >> 5] = bi; }
    __syncthreads();
    if (tid < 32) {
      best = s_val[tid]; bi = s_idx[tid];
#pragma unroll
      for (int off = 16; off > 0; off >>= 1) {
        float ov = __shfl_down(best, off, 32);
        int   oi = __shfl_down(bi,   off, 32);
        if (ov > best || (ov == best && oi < bi)) { best = ov; bi = oi; }
      }
      if (tid == 0) s_far = bi;
    }
    __syncthreads();
    far = s_far;
  }
}

// ---------------------------------------------------------------------------
// Ball query + gather: one wave32 per centroid (8 waves / 256-thread block).
// Order-preserving compaction via ballot+popcount (first 32 in-radius
// indices, pad with first). Writes X16 column-major [col][32ch] (f16) and
// new_xyz directly into d_out (B,3,S).
// ---------------------------------------------------------------------------
__global__ __launch_bounds__(256) void group_kernel(const float* __restrict__ xyz,
                                                    const float* __restrict__ pts,
                                                    const int* __restrict__ fidx,
                                                    _Float16* __restrict__ X16,
                                                    float* __restrict__ out_xyz) {
  int lane = threadIdx.x & 31, w = threadIdx.x >> 5;
  int gwave = blockIdx.x * 8 + w;
  int b = gwave >> 11;          // / NS
  int s = gwave & (NS - 1);
  const float* xb = xyz + b * 3 * NP;
  int fi = fidx[b * NS + s];
  float cx = xb[fi], cy = xb[NP + fi], cz = xb[2 * NP + fi];
  if (lane == 0) {
    out_xyz[b * 3 * NS + s]          = cx;
    out_xyz[b * 3 * NS + NS + s]     = cy;
    out_xyz[b * 3 * NS + 2 * NS + s] = cz;
  }
  float cn = cx * cx + cy * cy + cz * cz;
  __shared__ int s_nb[8][NK];
  int filled = 0;
  for (int base = 0; base < NP; base += 32) {
    int i = base + lane;
    float qx = xb[i], qy = xb[NP + i], qz = xb[2 * NP + i];
    // same formula as reference: |c|^2 + |p|^2 - 2 c.p
    float sq = cn + (qx * qx + qy * qy + qz * qz)
                  - 2.0f * (cx * qx + cy * qy + cz * qz);
    bool pred = sq <= R2;
    unsigned mask = (unsigned)__ballot(pred);
    int before = __popc(mask & ((1u << lane) - 1u));
    int pos = filled + before;
    if (pred && pos < NK) s_nb[w][pos] = i;
    filled += __popc(mask);
    if (filled >= NK) break;
  }
  __builtin_amdgcn_wave_barrier();
  if (filled > NK) filled = NK;
  int first = s_nb[w][0];           // centroid itself always qualifies
  if (lane >= filled) s_nb[w][lane] = first;
  __builtin_amdgcn_wave_barrier();
  int idx = s_nb[w][lane];          // lane == sample k
  int col = (b * NK + lane) * NS + s;
  _Float16* xc = X16 + (size_t)col * CINP;
  xc[0] = (_Float16)(xb[idx] - cx);
  xc[1] = (_Float16)(xb[NP + idx] - cy);
  xc[2] = (_Float16)(xb[2 * NP + idx] - cz);
  const float* pb = pts + (size_t)b * NDF * NP;
#pragma unroll
  for (int d = 0; d < NDF; d++) xc[3 + d] = (_Float16)pb[d * NP + idx];
}

// ---------------------------------------------------------------------------
// WMMA GEMM: Y[col][64] = W(64xCin) * X[col][Cin]   (f16 in, f32 acc, f16 out)
// One wave owns the FULL 64 output rows for 16 columns: the B fragment is
// loaded once per K-step and reused by 4 WMMAs (4 m-tiles) -> activation
// traffic 1x instead of 4x. 8 waves/block = 128 columns/block.
// A / B fragments use contiguous k (16*hi + s) in both; C follows the CDNA5 ISA 32-bit C table.
// ---------------------------------------------------------------------------
__device__ inline v16h load_afrag(const _Float16* __restrict__ W16, int cin,
                                  int row, int kk, int hi) {
  // lane<16: K kk+0..15 in a[0..15]
  // lane>=16: K kk+16..31
  const _Float16* wr = W16 + row * cin + kk + hi * 16;
  v16h a = *(const v16h*)wr;
  return a;
}

__device__ inline void store_ctile(_Float16* __restrict__ yc, v8f acc) {
  v8h o;
#pragma unroll
  for (int v = 0; v < 8; v++) o[v] = (_Float16)acc[v];
  *(v8h*)yc = o;
}

template <int CIN>
__global__ __launch_bounds__(256) void gemm_kernel(const _Float16* __restrict__ W16,
                                                   const _Float16* __restrict__ X,
                                                   _Float16* __restrict__ Y) {
  int lane = threadIdx.x & 31, w = threadIdx.x >> 5;
  int colBase = blockIdx.x * 128 + w * 16;
  int r = lane & 15, hi = lane >> 4;
  v8f acc0 = {}, acc1 = {}, acc2 = {}, acc3 = {};
#pragma unroll
  for (int kk = 0; kk < CIN; kk += 32) {
    // B (32x16): lane<16 -> column colBase+r, K kk+0..15; lane>=16 -> K kk+16..31
    const _Float16* xc = X + (size_t)(colBase + r) * CIN + kk + hi * 16;
    v16h bfr = *(const v16h*)xc;
    v16h a0 = load_afrag(W16, CIN,  0 + r, kk, hi);
    v16h a1 = load_afrag(W16, CIN, 16 + r, kk, hi);
    v16h a2 = load_afrag(W16, CIN, 32 + r, kk, hi);
    v16h a3 = load_afrag(W16, CIN, 48 + r, kk, hi);
    acc0 = __builtin_amdgcn_wmma_f32_16x16x32_f16(false, a0, false, bfr,
                                                  (short)0, acc0, false, false);
    acc1 = __builtin_amdgcn_wmma_f32_16x16x32_f16(false, a1, false, bfr,
                                                  (short)0, acc1, false, false);
    acc2 = __builtin_amdgcn_wmma_f32_16x16x32_f16(false, a2, false, bfr,
                                                  (short)0, acc2, false, false);
    acc3 = __builtin_amdgcn_wmma_f32_16x16x32_f16(false, a3, false, bfr,
                                                  (short)0, acc3, false, false);
  }
  // C: lane<16 holds M=mBase+0..7 of column colBase+r; lane>=16 M=mBase+8..15
  _Float16* yc = Y + (size_t)(colBase + r) * CMID + hi * 8;
  store_ctile(yc,      acc0);
  store_ctile(yc + 16, acc1);
  store_ctile(yc + 32, acc2);
  store_ctile(yc + 48, acc3);
}

// ---------------------------------------------------------------------------
// Per-channel sum / sum^2 over all NCOL columns (for batchnorm stats).
// ---------------------------------------------------------------------------
__global__ __launch_bounds__(256) void stats_kernel(const _Float16* __restrict__ Y,
                                                    float* __restrict__ sums,
                                                    float* __restrict__ sumsq) {
  int c = threadIdx.x & 63, cl = threadIdx.x >> 6;
  size_t col0 = (size_t)blockIdx.x * 256;
  float s = 0.0f, s2 = 0.0f;
  for (int i = cl; i < 256; i += 4) {
    float v = (float)Y[(col0 + i) * CMID + c];
    s += v; s2 += v * v;
  }
  __shared__ float sh[256], sh2[256];
  sh[threadIdx.x] = s; sh2[threadIdx.x] = s2;
  __syncthreads();
  if (cl == 0) {
    s  = sh[c]  + sh[64 + c]  + sh[128 + c]  + sh[192 + c];
    s2 = sh2[c] + sh2[64 + c] + sh2[128 + c] + sh2[192 + c];
    atomicAdd(&sums[c], s);
    atomicAdd(&sumsq[c], s2);
  }
}

__global__ void zero_kernel(float* p) { p[threadIdx.x] = 0.0f; }

__global__ void bnfin_kernel(const float* __restrict__ sums,
                             const float* __restrict__ sumsq,
                             const float* __restrict__ g,
                             const float* __restrict__ bb,
                             float* __restrict__ scale,
                             float* __restrict__ shift) {
  int c = threadIdx.x;
  float inv = 1.0f / (float)NCOL;
  float mean = sums[c] * inv;
  float var  = sumsq[c] * inv - mean * mean;
  float sc = g[c] * rsqrtf(var + BNEPS);
  scale[c] = sc;
  shift[c] = bb[c] - mean * sc;
}

// BN + optional residual + relu -> next f16 activation
__global__ __launch_bounds__(256) void apply_kernel(const _Float16* __restrict__ Y,
                                                    const float* __restrict__ scale,
                                                    const float* __restrict__ shift,
                                                    const _Float16* __restrict__ res,
                                                    _Float16* __restrict__ out) {
  size_t i = ((size_t)blockIdx.x * 256 + threadIdx.x) * 4;
#pragma unroll
  for (int j = 0; j < 4; j++) {
    int c = (int)((i + j) & 63);
    float v = (float)Y[i + j] * scale[c] + shift[c];
    if (res) v += (float)res[i + j];
    v = fmaxf(v, 0.0f);
    out[i + j] = (_Float16)v;
  }
}

// feat[b][c][s] = max_k act[col(b,k,s)][c]
__global__ __launch_bounds__(256) void maxk_kernel(const _Float16* __restrict__ X,
                                                   float* __restrict__ out) {
  int t = blockIdx.x * 256 + threadIdx.x;
  int c = t & 63;
  int rest = t >> 6;
  int s = rest & (NS - 1);
  int b = rest >> 11;
  float m = -3.4e38f;
#pragma unroll
  for (int k = 0; k < NK; k++) {
    float v = (float)X[(size_t)((b * NK + k) * NS + s) * CMID + c];
    m = fmaxf(m, v);
  }
  out[(size_t)(b * CMID + c) * NS + s] = m;
}

__global__ void cvtw_kernel(const float* __restrict__ w, _Float16* __restrict__ o, int n) {
  int i = blockIdx.x * 256 + threadIdx.x;
  if (i < n) o[i] = (_Float16)w[i];
}

// ---------------------------------------------------------------------------
extern "C" void kernel_launch(void* const* d_in, const int* in_sizes, int n_in,
                              void* d_out, int out_size, void* d_ws, size_t ws_size,
                              hipStream_t stream) {
  (void)in_sizes; (void)n_in; (void)out_size; (void)ws_size;
  const float* xyz    = (const float*)d_in[0];
  const float* pts    = (const float*)d_in[1];
  const float* proj_w = (const float*)d_in[2];
  const float* proj_g = (const float*)d_in[3];
  const float* proj_b = (const float*)d_in[4];
  const float* w1     = (const float*)d_in[5];
  const float* g1     = (const float*)d_in[6];
  const float* b1     = (const float*)d_in[7];
  const float* w2     = (const float*)d_in[8];
  const float* g2     = (const float*)d_in[9];
  const float* b2     = (const float*)d_in[10];
  float* out = (float*)d_out;

  // workspace layout (~118 MB total)
  char* ws = (char*)d_ws;
  int*   fidx  = (int*)ws;                          // 32768 B
  float* sums  = (float*)(ws + 32768);              // 64 f32
  float* sumsq = (float*)(ws + 33024);              // 64 f32 (contiguous w/ sums)
  float* scale = (float*)(ws + 33280);
  float* shift = (float*)(ws + 33536);
  const size_t X16_OFF = 65536;
  const size_t Y_OFF   = X16_OFF + (size_t)NCOL * CINP * 2;   // +16.8 MB
  const size_t A0_OFF  = Y_OFF   + (size_t)NCOL * CMID * 2;   // +33.5 MB
  const size_t A1_OFF  = A0_OFF  + (size_t)NCOL * CMID * 2;   // +33.5 MB
  const size_t W_OFF   = A1_OFF  + (size_t)NCOL * CMID * 2;   // +33.5 MB
  _Float16* X16  = (_Float16*)(ws + X16_OFF);
  _Float16* Yb   = (_Float16*)(ws + Y_OFF);
  _Float16* A0   = (_Float16*)(ws + A0_OFF);
  _Float16* A1   = (_Float16*)(ws + A1_OFF);
  _Float16* Wp16 = (_Float16*)(ws + W_OFF);
  _Float16* W116 = Wp16 + 2048;   // 2 x 64 x 64
  _Float16* W216 = W116 + 8192;

  cvtw_kernel<<<8,  256, 0, stream>>>(proj_w, Wp16, 2048);
  cvtw_kernel<<<32, 256, 0, stream>>>(w1, W116, 8192);
  cvtw_kernel<<<32, 256, 0, stream>>>(w2, W216, 8192);

  fps_kernel<<<NB, 1024, 0, stream>>>(xyz, fidx);
  group_kernel<<<NB * NS / 8, 256, 0, stream>>>(xyz, pts, fidx, X16, out);

  auto pass = [&](const _Float16* Wm, const _Float16* Xin, int cin,
                  const float* g, const float* bb,
                  const _Float16* res, _Float16* Aout) {
    if (cin == CINP)
      gemm_kernel<CINP><<<NCOL / 128, 256, 0, stream>>>(Wm, Xin, Yb);
    else
      gemm_kernel<CMID><<<NCOL / 128, 256, 0, stream>>>(Wm, Xin, Yb);
    zero_kernel<<<1, 128, 0, stream>>>(sums);                 // sums+sumsq
    stats_kernel<<<NCOL / 256, 256, 0, stream>>>(Yb, sums, sumsq);
    bnfin_kernel<<<1, 64, 0, stream>>>(sums, sumsq, g, bb, scale, shift);
    apply_kernel<<<(size_t)NCOL * CMID / 1024, 256, 0, stream>>>(Yb, scale, shift, res, Aout);
  };

  // projection conv + bn + relu
  pass(Wp16, X16, CINP, proj_g, proj_b, nullptr, A0);
  // residual block d = 0
  pass(W116,        A0, CMID, g1,      b1,      nullptr, A1);  // t = relu(bn(w1 x))
  pass(W216,        A1, CMID, g2,      b2,      A0,      A1);  // x = relu(bn(w2 t) + idn)
  // residual block d = 1
  pass(W116 + 4096, A1, CMID, g1 + 64, b1 + 64, nullptr, A0);
  pass(W216 + 4096, A0, CMID, g2 + 64, b2 + 64, A1,      A0);

  maxk_kernel<<<NB * CMID * NS / 256, 256, 0, stream>>>(A0, out + NB * 3 * NS);
}
